// SiglipAttention_58686433132811
// MI455X (gfx1250) — matrix operand order repaired and checked by reading, not compiled, not run on hardware
//
#include <hip/hip_runtime.h>
#include <stdint.h>

// ---------------------------------------------------------------------------
// Types for CDNA5 WMMA (wave32, v_wmma_f32_16x16x32_bf16)
// ---------------------------------------------------------------------------
typedef __attribute__((ext_vector_type(16))) __bf16 v16bf;
typedef __attribute__((ext_vector_type(8)))  __bf16 v8bf;
typedef __attribute__((ext_vector_type(8)))  float  v8f;

__device__ __forceinline__ v8f wmma_bf16(v16bf a, v16bf b, v8f c) {
  return __builtin_amdgcn_wmma_f32_16x16x32_bf16(
      /*neg_a=*/false, a, /*neg_b=*/false, b,
      /*c_mod=*/(short)0, c, /*reuse_a=*/false, /*reuse_b=*/false);
}

// A-fragment: lane holds row M=lane&15; 16 bf16 = K {k0..k0+7} and {16+k0..},
// k0 = (lane>=16)*8  (ISA 16-bit A-matrix table, wave32)
__device__ __forceinline__ v16bf load_frag_a(const __bf16* row, int k0) {
  v8bf lo = *(const v8bf*)(row + k0);
  v8bf hi = *(const v8bf*)(row + 16 + k0);
  v16bf f;
#pragma unroll
  for (int i = 0; i < 8; ++i) { f[i] = lo[i]; f[i + 8] = hi[i]; }
  return f;
}

// B-fragment (rows of B^T): lane holds column N=lane&15; 16 bf16 = K {k0..k0+7}
// and {16+k0..}, k0 = kb/2 = (lane>=16)*8 (same order as A-fragment)
__device__ __forceinline__ v16bf load_frag_b(const __bf16* row, int kb) {
  v8bf lo = *(const v8bf*)(row + kb / 2);
  v8bf hi = *(const v8bf*)(row + 16 + kb / 2);
  v16bf f;
#pragma unroll
  for (int i = 0; i < 8; ++i) { f[i] = lo[i]; f[i + 8] = hi[i]; }
  return f;
}

// ---------------------------------------------------------------------------
// fp32 -> bf16 conversion
// ---------------------------------------------------------------------------
__global__ void cvt_f32_to_bf16(const float* __restrict__ in,
                                __bf16* __restrict__ out, int n) {
  int i = (blockIdx.x * blockDim.x + threadIdx.x) * 4;
  if (i + 3 < n) {
    float4 v = *(const float4*)(in + i);
    union { __bf16 b[4]; uint2 u; } o;
    o.b[0] = (__bf16)v.x; o.b[1] = (__bf16)v.y;
    o.b[2] = (__bf16)v.z; o.b[3] = (__bf16)v.w;
    *(uint2*)(out + i) = o.u;
  }
}

// ---------------------------------------------------------------------------
// GEMM: C[M,N] = (A[M,K](bf16) * W[N,K]^T(bf16) + bias[N](f32)) * outScale
// 128x128 tile, 8 waves in 4(m) x 2(n) grid; wave = 32x64 strip
// (2 A-frags x 4 B-frags -> 8 WMMAs per 32-wide K step).
// Software pipelined: next K-step's global loads issue before this step's
// WMMAs; +prefetch (global_prefetch_b8) one further step ahead.
// TRANSPOSE_OUT: bf16 output as [B][N][T] with packed 16B stores.
// ---------------------------------------------------------------------------
template <typename OutT, bool TRANSPOSE_OUT>
__global__ __launch_bounds__(256)
void gemm_bias_kernel(const __bf16* __restrict__ A, const __bf16* __restrict__ W,
                      const float* __restrict__ bias, OutT* __restrict__ Cout,
                      int M, int N, int K, int Trows, float outScale) {
  __shared__ __bf16 As[128][40];  // 80B rows, 16B aligned
  __shared__ __bf16 Bs[128][40];

  const int tid  = threadIdx.x;
  const int wave = tid >> 5;
  const int lane = tid & 31;
  const int wm = wave & 3;    // rows wm*32
  const int wn = wave >> 2;   // cols wn*64
  const int m0 = blockIdx.y * 128;
  const int n0 = blockIdx.x * 128;

  const int am   = lane & 15;
  const int half = lane >> 4;
  const int k0 = half * 8;
  const int kb = half * 16;
  const int lrow = tid >> 1;        // 0..127
  const int lcol = (tid & 1) * 16;  // 0 / 16

  v8f acc[2][4] = {};

  const __bf16* gA = A + (size_t)(m0 + lrow) * K + lcol;
  const __bf16* gW = W + (size_t)(n0 + lrow) * K + lcol;

  // stage 0 into registers
  v8bf ra0 = *(const v8bf*)(gA);
  v8bf ra1 = *(const v8bf*)(gA + 8);
  v8bf rb0 = *(const v8bf*)(gW);
  v8bf rb1 = *(const v8bf*)(gW + 8);

  for (int k = 0; k < K; k += 32) {
    __syncthreads();
    *(v8bf*)&As[lrow][lcol]     = ra0;
    *(v8bf*)&As[lrow][lcol + 8] = ra1;
    *(v8bf*)&Bs[lrow][lcol]     = rb0;
    *(v8bf*)&Bs[lrow][lcol + 8] = rb1;
    __syncthreads();

    if (k + 32 < K) {
      ra0 = *(const v8bf*)(gA + k + 32);
      ra1 = *(const v8bf*)(gA + k + 40);
      rb0 = *(const v8bf*)(gW + k + 32);
      rb1 = *(const v8bf*)(gW + k + 40);
      if (k + 64 < K) {
        __builtin_prefetch(gA + k + 64, 0, 0);
        __builtin_prefetch(gW + k + 64, 0, 0);
      }
    }

    v16bf af0 = load_frag_a(&As[wm * 32 + am][0], k0);
    v16bf af1 = load_frag_a(&As[wm * 32 + 16 + am][0], k0);
#pragma unroll
    for (int c = 0; c < 4; ++c) {
      v16bf bf_ = load_frag_b(&Bs[wn * 64 + c * 16 + am][0], kb);
      acc[0][c] = wmma_bf16(af0, bf_, acc[0][c]);
      acc[1][c] = wmma_bf16(af1, bf_, acc[1][c]);
    }
  }

#pragma unroll
  for (int mg = 0; mg < 2; ++mg) {
#pragma unroll
    for (int c = 0; c < 4; ++c) {
      const int n = n0 + wn * 64 + c * 16 + am;
      const float bsc = bias[n];
      if constexpr (TRANSPOSE_OUT) {
        // 8 consecutive m at fixed n -> one packed row-segment of [B][N][T]
        const int mbase = m0 + wm * 32 + mg * 16 + half * 8;
        const int bidx  = mbase / Trows;
        const int t     = mbase % Trows;
        union { __bf16 b8[8]; v8bf v; } pk;
#pragma unroll
        for (int r = 0; r < 8; ++r)
          pk.b8[r] = (__bf16)((acc[mg][c][r] + bsc) * outScale);
        __bf16* Ct = (__bf16*)Cout;
        *(v8bf*)&Ct[((size_t)bidx * N + n) * Trows + t] = pk.v;
      } else {
#pragma unroll
        for (int r = 0; r < 8; ++r) {
          const int m = m0 + wm * 32 + mg * 16 + r + half * 8;
          Cout[(size_t)m * N + n] = (OutT)((acc[mg][c][r] + bsc) * outScale);
        }
      }
    }
  }
}

// ---------------------------------------------------------------------------
// Flash attention: workgroup = 64 query rows of one (b,h); 4 waves.
// Q is pre-scaled by 1/sqrt(D) in its projection, so no per-tile scaling.
// K staged [key][d]; V staged from pre-transposed Vt [B][C][T] as [d][key].
// ---------------------------------------------------------------------------
__global__ __launch_bounds__(128)
void flash_attn_kernel(const __bf16* __restrict__ Qg, const __bf16* __restrict__ Kg,
                       const __bf16* __restrict__ Vt, __bf16* __restrict__ Og,
                       int T, int Cdim, int Dd) {
  __shared__ __bf16 Ks[64][72];   // [key][d]  144B rows, 16B aligned
  __shared__ __bf16 Vts[64][72];  // [d][key]
  __shared__ __bf16 Ps[64][72];   // [q][key]  wave-private row blocks

  const int tid  = threadIdx.x;
  const int wave = tid >> 5;
  const int lane = tid & 31;
  const int am   = lane & 15;
  const int half = lane >> 4;
  const int k0 = half * 8;
  const int kb = half * 16;

  const int qt0 = blockIdx.x * 64;
  const int h   = blockIdx.y;
  const int b   = blockIdx.z;
  const size_t baseRow = (size_t)b * T;
  const int hoff = h * Dd;

  const __bf16* qrow = Qg + (baseRow + qt0 + wave * 16 + am) * Cdim + hoff;
  const v16bf aq0 = load_frag_a(qrow, k0);
  const v16bf aq1 = load_frag_a(qrow + 32, k0);

  float m_[8], l_[8];
#pragma unroll
  for (int r = 0; r < 8; ++r) { m_[r] = -1e30f; l_[r] = 0.f; }
  v8f oacc[4] = {};

  const int lrow  = tid >> 1;        // 0..63
  const int lcolh = (tid & 1) * 32;  // 0 / 32

  const __bf16* gkBase = Kg + (baseRow + lrow) * Cdim + hoff + lcolh;
  const __bf16* gvBase = Vt + ((size_t)b * Cdim + hoff + lrow) * T + lcolh;

  for (int kt = 0; kt < T; kt += 64) {
    __syncthreads();
    {
      // K tile rows [key][d]
      const __bf16* gk = gkBase + (size_t)kt * Cdim;
#pragma unroll
      for (int j = 0; j < 4; ++j)
        *(v8bf*)&Ks[lrow][lcolh + j * 8] = *(const v8bf*)(gk + j * 8);
      // V^T tile rows [d][key]
      const __bf16* gv = gvBase + kt;
#pragma unroll
      for (int j = 0; j < 4; ++j)
        *(v8bf*)&Vts[lrow][lcolh + j * 8] = *(const v8bf*)(gv + j * 8);
      // prefetch next key tile (global_prefetch_b8)
      if (kt + 64 < T) {
        __builtin_prefetch(gkBase + (size_t)(kt + 64) * Cdim, 0, 0);
        __builtin_prefetch(gvBase + kt + 64, 0, 0);
      }
    }
    __syncthreads();

    // S = Q K^T  (Q carries the 1/sqrt(D) scale)
    v8f sacc[4] = {};
#pragma unroll
    for (int c = 0; c < 4; ++c) {
      v16bf kf0 = load_frag_b(&Ks[c * 16 + am][0], kb);
      sacc[c] = wmma_bf16(aq0, kf0, sacc[c]);
      v16bf kf1 = load_frag_b(&Ks[c * 16 + am][32], kb);
      sacc[c] = wmma_bf16(aq1, kf1, sacc[c]);
    }

    // online softmax
    float rmax[8];
#pragma unroll
    for (int r = 0; r < 8; ++r)
      rmax[r] = fmaxf(fmaxf(sacc[0][r], sacc[1][r]), fmaxf(sacc[2][r], sacc[3][r]));
#pragma unroll
    for (int mask = 1; mask < 16; mask <<= 1)
#pragma unroll
      for (int r = 0; r < 8; ++r)
        rmax[r] = fmaxf(rmax[r], __shfl_xor(rmax[r], mask, 32));

    float newm[8], alpha[8], rsum[8];
#pragma unroll
    for (int r = 0; r < 8; ++r) {
      newm[r]  = fmaxf(m_[r], rmax[r]);
      alpha[r] = __expf(m_[r] - newm[r]);
      rsum[r]  = 0.f;
    }
#pragma unroll
    for (int c = 0; c < 4; ++c)
#pragma unroll
      for (int r = 0; r < 8; ++r) {
        float p = __expf(sacc[c][r] - newm[r]);
        rsum[r] += p;
        Ps[wave * 16 + r + half * 8][c * 16 + am] = (__bf16)p;
      }
#pragma unroll
    for (int mask = 1; mask < 16; mask <<= 1)
#pragma unroll
      for (int r = 0; r < 8; ++r) rsum[r] += __shfl_xor(rsum[r], mask, 32);
#pragma unroll
    for (int r = 0; r < 8; ++r) {
      l_[r] = l_[r] * alpha[r] + rsum[r];
      m_[r] = newm[r];
    }
#pragma unroll
    for (int c = 0; c < 4; ++c)
#pragma unroll
      for (int r = 0; r < 8; ++r) oacc[c][r] *= alpha[r];

    // O += P V   (Ps rows are wave-private; LDS ops in-order per wave)
    v16bf pf0 = load_frag_a(&Ps[wave * 16 + am][0], k0);
    v16bf pf1 = load_frag_a(&Ps[wave * 16 + am][32], k0);
#pragma unroll
    for (int c = 0; c < 4; ++c) {
      v16bf vf0 = load_frag_b(&Vts[c * 16 + am][0], kb);
      oacc[c] = wmma_bf16(pf0, vf0, oacc[c]);
      v16bf vf1 = load_frag_b(&Vts[c * 16 + am][32], kb);
      oacc[c] = wmma_bf16(pf1, vf1, oacc[c]);
    }
  }

#pragma unroll
  for (int c = 0; c < 4; ++c)
#pragma unroll
    for (int r = 0; r < 8; ++r) {
      float v = oacc[c][r] / l_[r];
      size_t row = baseRow + qt0 + wave * 16 + r + half * 8;
      Og[row * Cdim + hoff + c * 16 + am] = (__bf16)v;
    }
}

// ---------------------------------------------------------------------------
// Host-side orchestration
// ---------------------------------------------------------------------------
extern "C" void kernel_launch(void* const* d_in, const int* in_sizes, int n_in,
                              void* d_out, int out_size, void* d_ws, size_t ws_size,
                              hipStream_t stream) {
  (void)in_sizes; (void)n_in; (void)out_size; (void)ws_size;
  const int Bv = 32, Tv = 1024, Cv = 768, Hv = 12, Dv = 64;
  const size_t MT = (size_t)Bv * Tv;   // 32768 rows
  const size_t nX = MT * Cv;           // 25,165,824 elems
  const size_t nW = (size_t)Cv * Cv;   // 589,824 elems

  const float* hs  = (const float*)d_in[0];
  const float* qw  = (const float*)d_in[1];
  const float* qb  = (const float*)d_in[2];
  const float* kw  = (const float*)d_in[3];
  const float* kb_ = (const float*)d_in[4];
  const float* vw  = (const float*)d_in[5];
  const float* vb  = (const float*)d_in[6];
  const float* ow  = (const float*)d_in[7];
  const float* ob  = (const float*)d_in[8];

  char* ws = (char*)d_ws;
  __bf16* Xb = (__bf16*)ws;  ws += nX * 2;
  __bf16* Wq = (__bf16*)ws;  ws += nW * 2;
  __bf16* Wk = (__bf16*)ws;  ws += nW * 2;
  __bf16* Wv = (__bf16*)ws;  ws += nW * 2;
  __bf16* Wo = (__bf16*)ws;  ws += nW * 2;
  __bf16* Qb = (__bf16*)ws;  ws += nX * 2;
  __bf16* Kb = (__bf16*)ws;  ws += nX * 2;
  __bf16* Vt = (__bf16*)ws;  ws += nX * 2;  // [B][C][T] transposed V
  __bf16* Ab = (__bf16*)ws;  ws += nX * 2;

  // fp32 -> bf16
  {
    int bx = (int)((nX / 4 + 255) / 256);
    cvt_f32_to_bf16<<<bx, 256, 0, stream>>>(hs, Xb, (int)nX);
    int bw = (int)((nW / 4 + 255) / 256);
    cvt_f32_to_bf16<<<bw, 256, 0, stream>>>(qw, Wq, (int)nW);
    cvt_f32_to_bf16<<<bw, 256, 0, stream>>>(kw, Wk, (int)nW);
    cvt_f32_to_bf16<<<bw, 256, 0, stream>>>(vw, Wv, (int)nW);
    cvt_f32_to_bf16<<<bw, 256, 0, stream>>>(ow, Wo, (int)nW);
  }

  // Projections: 128x128 tiles -> grid (768/128, 32768/128) = (6, 256)
  dim3 gg(Cv / 128, (unsigned)(MT / 128));
  // Q carries the softmax scale 1/sqrt(D) = 0.125
  gemm_bias_kernel<__bf16, false><<<gg, 256, 0, stream>>>(Xb, Wq, qb,  Qb, (int)MT, Cv, Cv, Tv, 0.125f);
  gemm_bias_kernel<__bf16, false><<<gg, 256, 0, stream>>>(Xb, Wk, kb_, Kb, (int)MT, Cv, Cv, Tv, 1.0f);
  // V projection written pre-transposed to [B][C][T]
  gemm_bias_kernel<__bf16, true ><<<gg, 256, 0, stream>>>(Xb, Wv, vb,  Vt, (int)MT, Cv, Cv, Tv, 1.0f);

  // Flash attention
  flash_attn_kernel<<<dim3(Tv / 64, Hv, Bv), 128, 0, stream>>>(
      Qb, Kb, Vt, Ab, Tv, Cv, Dv);

  // Output projection (fp32 out)
  gemm_bias_kernel<float, false><<<gg, 256, 0, stream>>>(Ab, Wo, ob, (float*)d_out, (int)MT, Cv, Cv, Tv, 1.0f);
}
